// NeuralMeshSimplification_66400194396558
// MI455X (gfx1250) — matrix operand order repaired and checked by reading, not compiled, not run on hardware
//
#include <hip/hip_runtime.h>
#include <hip/hip_fp16.h>
#include <math.h>

typedef __attribute__((ext_vector_type(2)))  float    v2f;
typedef __attribute__((ext_vector_type(8)))  float    v8f;
typedef __attribute__((ext_vector_type(16))) _Float16 v16h;

#define NN      8192
#define FD      64
#define EE      262144
#define KTOP    16
#define NPAIR   120          // K*(K-1)/2
#define NTRI    (NN * NPAIR) // 983040

// ---- d_out layout (float elements, concatenated in reference return order)
#define OUT_TRI    0               // triangles  (NTRI*3)
#define OUT_TP     (NTRI * 3)      // tri_probs  (NTRI)
#define OUT_VALID  (NTRI * 4)      // valid      (NTRI)
#define OUT_FEATS  (NTRI * 5)      // feats      (NTRI*64)
#define OUT_CENTS  (NTRI * 69)     // cents      (NTRI*3)
#define OUT_KEEP   (NTRI * 72)     // keep       (NTRI)   total NTRI*73

// ---- workspace layout (int32 units). First WS_ZERO_N ints are zeroed per call.
#define WS_ROWCNT   0
#define WS_CURSOR   (WS_ROWCNT + NN)
#define WS_HIST1    (WS_CURSOR + NN)
#define WS_HIST2A   (WS_HIST1 + 65536)
#define WS_HIST2B   (WS_HIST2A + 65536)
#define WS_SCAL     (WS_HIST2B + 65536)
#define WS_ZERO_N   (WS_SCAL + 16)
#define WS_ROWSTART (WS_ZERO_N)
#define WS_CSRCOL   (WS_ROWSTART + NN + 1)
#define WS_CSRVAL   (WS_CSRCOL + EE)
#define WS_TOPI     (WS_CSRVAL + EE)
#define WS_TOPV     (WS_TOPI + NN * KTOP)
// end ~= 4.03 MB of scratch

// scalars: [0]=nv [1]=binLo [2]=binHi [3]=remLo [4]=remHi [5]=w(f32) [8]=thr(f32)

// ---------------------------------------------------------------- CSR build
__global__ void count_kernel(const int* __restrict__ ei, int* __restrict__ rowcnt) {
  int e = blockIdx.x * blockDim.x + threadIdx.x;
  if (e < EE) atomicAdd(&rowcnt[ei[e]], 1);
}

__global__ void scan_kernel(const int* __restrict__ rowcnt, int* __restrict__ rowstart) {
  __shared__ int part[256];
  const int tid = threadIdx.x;
  const int C = NN / 256; // 32 rows per thread
  const int base = tid * C;
  int s = 0;
  for (int i = 0; i < C; ++i) s += rowcnt[base + i];
  part[tid] = s;
  __syncthreads();
  for (int off = 1; off < 256; off <<= 1) {   // Hillis-Steele inclusive scan
    int v = (tid >= off) ? part[tid - off] : 0;
    __syncthreads();
    part[tid] += v;
    __syncthreads();
  }
  int run = part[tid] - s;                    // exclusive prefix for this chunk
  for (int i = 0; i < C; ++i) { rowstart[base + i] = run; run += rowcnt[base + i]; }
  if (tid == 255) rowstart[NN] = run;         // == EE
}

__global__ void scatter_kernel(const int* __restrict__ ei, const float* __restrict__ ep,
                               const int* __restrict__ rowstart, int* __restrict__ cursor,
                               int* __restrict__ col, float* __restrict__ val) {
  int e = blockIdx.x * blockDim.x + threadIdx.x;
  if (e < EE) {
    int r = ei[e], c = ei[EE + e];
    int ofs = atomicAdd(&cursor[r], 1);
    int idx = rowstart[r] + ofs;
    col[idx] = c;
    val[idx] = ep[e];
  }
}

// ---------------------------------------------------------------- top-16 per row
__global__ void topk_kernel(const int* __restrict__ rowstart, const int* __restrict__ col,
                            const float* __restrict__ val,
                            int* __restrict__ topi, float* __restrict__ topv) {
  int r = blockIdx.x * blockDim.x + threadIdx.x;
  if (r >= NN) return;
  float bv[KTOP]; int bi[KTOP];
  for (int k = 0; k < KTOP; ++k) { bv[k] = -1.0f; bi[k] = k; }
  const int beg = rowstart[r], end = rowstart[r + 1];
  for (int e = beg; e < end; ++e) {
    float v = val[e];
    if (v > bv[KTOP - 1]) {
      int c = col[e];
      int k = KTOP - 1;
      while (k > 0 && bv[k - 1] < v) { bv[k] = bv[k - 1]; bi[k] = bi[k - 1]; --k; }
      bv[k] = v; bi[k] = c;
    }
  }
  for (int k = 0; k < KTOP; ++k) {
    topv[r * KTOP + k] = bv[k] < 0.f ? 0.f : bv[k];  // pad rows with deg<16
    topi[r * KTOP + k] = bi[k];
  }
}

// ---------------------------------------------------------------- main triangle kernel
#define WAVES 8

__device__ __forceinline__ void pair_decode(int t, int& j, int& l) {
  // t-th pair of jnp.triu_indices(16,1): (0,1)(0,2)...(0,15)(1,2)...
  int jj = 0, rem = t, cnt = KTOP - 1;
  while (rem >= cnt) { rem -= cnt; ++jj; --cnt; }
  j = jj; l = jj + 1 + rem;
}

__global__ void __launch_bounds__(256)
tri_kernel(const float* __restrict__ x, const float* __restrict__ pos,
           const int* __restrict__ rowstart, const int* __restrict__ col,
           const float* __restrict__ val,
           const int* __restrict__ topi, const float* __restrict__ topv,
           float* __restrict__ out)
{
  __shared__ float xg[WAVES][32 * FD];     // Xg: row0=center, 1..16=neighbors, 17..31=0
  __shared__ float scale_s[WAVES][128];    // per-triangle 0 or 1/3
  __shared__ int   ti_s[WAVES][KTOP];
  __shared__ float tv_s[WAVES][KTOP];

  const int lane = threadIdx.x & 31;
  const int wave = threadIdx.x >> 5;
  const int node = blockIdx.x * WAVES + wave;   // grid exactly covers NN
  const int li   = lane & 15;                   // wave32 WMMA: lane%16
  const int hs   = lane >> 4;                   // half-wave select

  if (lane < KTOP) {
    ti_s[wave][lane] = topi[node * KTOP + lane];
    tv_s[wave][lane] = topv[node * KTOP + lane];
  }
  __syncthreads();

  // Stage center + 16 neighbor feature rows into LDS; zero-pad K rows 17..31 so
  // every WMMA B-fragment load is unconditional (no EXEC divergence in the MMA loop).
  for (int r = 0; r < 17; ++r) {
    int src = (r == 0) ? node : ti_s[wave][r - 1];
    xg[wave][r * FD + lane]      = x[src * FD + lane];
    xg[wave][r * FD + lane + 32] = x[src * FD + lane + 32];
  }
  for (int r = 17; r < 32; ++r) {
    xg[wave][r * FD + lane]      = 0.f;
    xg[wave][r * FD + lane + 32] = 0.f;
  }

  // Per-pair phase: p3 probe in L2-resident CSR, tri prob, triangles/valid/cents.
  const float pcx = pos[node * 3 + 0], pcy = pos[node * 3 + 1], pcz = pos[node * 3 + 2];
  for (int t = lane; t < NPAIR; t += 32) {
    int j, l; pair_decode(t, j, l);
    const int   n1 = ti_s[wave][j], n2 = ti_s[wave][l];
    const float p1 = tv_s[wave][j], p2 = tv_s[wave][l];
    float p3 = 0.f;
    const int beg = rowstart[n1], end = rowstart[n1 + 1];
    for (int e = beg; e < end; ++e)
      if (col[e] == n2) { p3 = val[e]; break; }      // ~32 entries avg, L2-resident
    const bool  vld  = p3 > 0.f;
    const float prod = p1 * p2 * (vld ? p3 : 1.f);
    const float tp   = vld ? cbrtf(prod) : 0.f;
    const float s    = vld ? (1.f / 3.f) : 0.f;
    const int   ti   = node * NPAIR + t;
    out[OUT_TRI + ti * 3 + 0] = (float)node;
    out[OUT_TRI + ti * 3 + 1] = (float)n1;
    out[OUT_TRI + ti * 3 + 2] = (float)n2;
    out[OUT_TP + ti]          = tp;
    out[OUT_VALID + ti]       = vld ? 1.f : 0.f;
    scale_s[wave][t]          = s;
    out[OUT_CENTS + ti * 3 + 0] = s * (pcx + pos[n1 * 3 + 0] + pos[n2 * 3 + 0]);
    out[OUT_CENTS + ti * 3 + 1] = s * (pcy + pos[n1 * 3 + 1] + pos[n2 * 3 + 1]);
    out[OUT_CENTS + ti * 3 + 2] = s * (pcz + pos[n1 * 3 + 2] + pos[n2 * 3 + 2]);
  }
  for (int t = NPAIR + lane; t < 128; t += 32) scale_s[wave][t] = 0.f;
  __syncthreads();

  // feats = (diag(scale) . P)(128x32) @ Xg(32x64) on the matrix pipe.
  // The valid/3 scale is folded into the A matrix (A entries are 0 or scale[M]),
  // so D needs no post-processing. Kernel is store-BW bound; WMMA flops are free.
  const float* __restrict__ xw = xg[wave];
  const float* __restrict__ sw = scale_s[wave];

  #pragma unroll 1
  for (int mt = 0; mt < 8; ++mt) {
    const int M = mt * 16 + li;        // A-matrix row for this lane (both half-waves)
    int jjM = 127, llM = 127;
    if (M < NPAIR) pair_decode(M, jjM, llM);
    const float sM = sw[M];            // 0 or 1/3; 0 for rows >= 120 or invalid

#if __has_builtin(__builtin_amdgcn_wmma_f32_16x16x4_f32)
    // A fragments depend only on (mt,kt): cache across the 4 N-tiles.
    // Layout (ISA 7.12.2, 32-bit A 16x4): lanes0-15 K={0,1}, lanes16-31 K={2,3}.
    float ax[8], ay[8];
    #pragma unroll
    for (int kt = 0; kt < 8; ++kt) {
      const int k0 = kt * 4 + hs * 2;
      ax[kt] = (k0     == 0 || k0     == 1 + jjM || k0     == 1 + llM) ? sM : 0.f;
      ay[kt] = (k0 + 1 == 0 || k0 + 1 == 1 + jjM || k0 + 1 == 1 + llM) ? sM : 0.f;
    }
#endif

    #pragma unroll
    for (int nt = 0; nt < 4; ++nt) {
      v8f acc = {0.f, 0.f, 0.f, 0.f, 0.f, 0.f, 0.f, 0.f};
      const int ncol = nt * 16 + li;

#if __has_builtin(__builtin_amdgcn_wmma_f32_16x16x4_f32)
      // Full-precision f32 WMMA: D(16x16) += A(16x4) x B(4x16), 8 K-steps.
      #pragma unroll
      for (int kt = 0; kt < 8; ++kt) {
        const int k0 = kt * 4 + hs * 2;
        v2f a; a[0] = ax[kt]; a[1] = ay[kt];
        v2f b; b[0] = xw[k0 * FD + ncol]; b[1] = xw[(k0 + 1) * FD + ncol];
        acc = __builtin_amdgcn_wmma_f32_16x16x4_f32(false, a, false, b,
                                                    (short)0, acc, false, false);
      }
#elif __has_builtin(__builtin_amdgcn_wmma_f32_16x16x32_f16)
      // Fallback: one 16x16x32 f16 WMMA (approximate; scale folded the same way).
      v16h a, b;
      #pragma unroll
      for (int e = 0; e < 16; ++e) {
        const int Ka = hs * 16 + e;
        a[e] = (_Float16)((Ka == 0 || Ka == 1 + jjM || Ka == 1 + llM) ? sM : 0.f);
        const int Kb = hs * 16 + e;
        b[e] = (_Float16)xw[Kb * FD + ncol];
      }
      acc = __builtin_amdgcn_wmma_f32_16x16x32_f16(false, a, false, b,
                                                   (short)0, acc, false, false);
#else
      // Host-pass placeholder only (never executed on gfx1250).
      for (int k = 0; k < 32; ++k) {
        const float pv = (k == 0 || k == 1 + jjM || k == 1 + llM) ? sM : 0.f;
        acc[0] += pv * xw[k * FD + ncol];
      }
#endif

      // C/D layout: VGPR r holds rows mt*16 + hs*8 + r, cols = nt*16 + lane%16.
      // Only the last M-tile has rows >= 120: uniform branch for mt<7, one
      // predicated block (hs==0 -> rows 112..119) for mt==7.
      if (mt < 7) {
        const int tb = node * NPAIR + mt * 16 + hs * 8;
        #pragma unroll
        for (int r = 0; r < 8; ++r)
          out[OUT_FEATS + (tb + r) * FD + ncol] = acc[r];
      } else if (hs == 0) {
        const int tb = node * NPAIR + 112;
        #pragma unroll
        for (int r = 0; r < 8; ++r)
          out[OUT_FEATS + (tb + r) * FD + ncol] = acc[r];
      }
    }
  }
}

// ---------------------------------------------------------------- exact masked quantile
__global__ void hist1_kernel(const float* __restrict__ out, int* __restrict__ hist1) {
  int t = blockIdx.x * blockDim.x + threadIdx.x;
  if (t < NTRI && out[OUT_VALID + t] != 0.f) {
    unsigned key = __float_as_uint(out[OUT_TP + t]);  // probs >= 0 -> bits monotonic
    atomicAdd(&hist1[key >> 16], 1);
  }
}

__global__ void qscan1_kernel(const int* __restrict__ hist1, int* __restrict__ scal) {
  __shared__ int csum[256];
  __shared__ int shv[4]; // binLo, remLo, binHi, remHi
  const int tid = threadIdx.x;
  if (tid == 0) { shv[0] = 0; shv[1] = 0; shv[2] = 0; shv[3] = 0; }
  const int base = tid * 256;
  int s = 0;
  for (int i = 0; i < 256; ++i) s += hist1[base + i];
  csum[tid] = s;
  __syncthreads();
  for (int off = 1; off < 256; off <<= 1) {
    int v = (tid >= off) ? csum[tid - off] : 0;
    __syncthreads();
    csum[tid] += v;
    __syncthreads();
  }
  const int nv = csum[255];
  const float p = 0.5f * (float)(nv - 1);   // q*(nv-1), q = 1-TARGET_RATIO = 0.5
  const int lo = (nv > 0) ? (int)floorf(p) : 0;
  const int hi = (nv > 0) ? (int)ceilf(p) : 0;
  const int excl = csum[tid] - s;
  if (nv > 0) {
    if (lo >= excl && lo < csum[tid]) {
      int run = excl;
      for (int i = 0; i < 256; ++i) {
        int h = hist1[base + i];
        if (lo < run + h) { shv[0] = base + i; shv[1] = lo - run; break; }
        run += h;
      }
    }
    if (hi >= excl && hi < csum[tid]) {
      int run = excl;
      for (int i = 0; i < 256; ++i) {
        int h = hist1[base + i];
        if (hi < run + h) { shv[2] = base + i; shv[3] = hi - run; break; }
        run += h;
      }
    }
  }
  __syncthreads();
  if (tid == 0) {
    scal[0] = nv; scal[1] = shv[0]; scal[2] = shv[2]; scal[3] = shv[1]; scal[4] = shv[3];
    ((float*)scal)[5] = (nv > 0) ? (p - floorf(p)) : 0.f;
  }
}

__global__ void hist2_kernel(const float* __restrict__ out, const int* __restrict__ scal,
                             int* __restrict__ h2a, int* __restrict__ h2b) {
  int t = blockIdx.x * blockDim.x + threadIdx.x;
  if (t >= NTRI) return;
  if (out[OUT_VALID + t] == 0.f) return;
  const int bl = scal[1], bh = scal[2];
  const unsigned key = __float_as_uint(out[OUT_TP + t]);
  const int hi16 = (int)(key >> 16);
  if (hi16 == bl) atomicAdd(&h2a[key & 0xffffu], 1);
  if (hi16 == bh) atomicAdd(&h2b[key & 0xffffu], 1);
}

__global__ void qscan2_kernel(const int* __restrict__ h2a, const int* __restrict__ h2b,
                              int* __restrict__ scal) {
  __shared__ int csum[256];
  __shared__ unsigned shkey[2];
  const int tid = threadIdx.x;
  if (tid == 0) { shkey[0] = 0; shkey[1] = 0; }
  const int nv = scal[0];
  const int binLo = scal[1], binHi = scal[2], remLo = scal[3], remHi = scal[4];
  const float w = ((const float*)scal)[5];
  const int base = tid * 256;

  // select remLo-th within binLo
  int s = 0;
  for (int i = 0; i < 256; ++i) s += h2a[base + i];
  csum[tid] = s;
  __syncthreads();
  for (int off = 1; off < 256; off <<= 1) {
    int v = (tid >= off) ? csum[tid - off] : 0;
    __syncthreads();
    csum[tid] += v;
    __syncthreads();
  }
  int excl = csum[tid] - s;
  if (nv > 0 && remLo >= excl && remLo < csum[tid]) {
    int run = excl;
    for (int i = 0; i < 256; ++i) {
      int h = h2a[base + i];
      if (remLo < run + h) { shkey[0] = ((unsigned)binLo << 16) | (unsigned)(base + i); break; }
      run += h;
    }
  }
  __syncthreads();

  // select remHi-th within binHi
  s = 0;
  for (int i = 0; i < 256; ++i) s += h2b[base + i];
  csum[tid] = s;
  __syncthreads();
  for (int off = 1; off < 256; off <<= 1) {
    int v = (tid >= off) ? csum[tid - off] : 0;
    __syncthreads();
    csum[tid] += v;
    __syncthreads();
  }
  excl = csum[tid] - s;
  if (nv > 0 && remHi >= excl && remHi < csum[tid]) {
    int run = excl;
    for (int i = 0; i < 256; ++i) {
      int h = h2b[base + i];
      if (remHi < run + h) { shkey[1] = ((unsigned)binHi << 16) | (unsigned)(base + i); break; }
      run += h;
    }
  }
  __syncthreads();
  if (tid == 0) {
    float thr;
    if (nv > 0) {
      const float vLo = __uint_as_float(shkey[0]);
      const float vHi = __uint_as_float(shkey[1]);
      thr = (1.f - w) * vLo + w * vHi;
    } else {
      thr = 3.4e38f;
    }
    ((float*)scal)[8] = thr;
  }
}

__global__ void keep_kernel(const int* __restrict__ scal, float* __restrict__ out) {
  int t = blockIdx.x * blockDim.x + threadIdx.x;
  if (t >= NTRI) return;
  const float thr = ((const float*)scal)[8];
  out[OUT_KEEP + t] = (out[OUT_VALID + t] != 0.f && out[OUT_TP + t] > thr) ? 1.f : 0.f;
}

// ---------------------------------------------------------------- launch
extern "C" void kernel_launch(void* const* d_in, const int* in_sizes, int n_in,
                              void* d_out, int out_size, void* d_ws, size_t ws_size,
                              hipStream_t stream) {
  (void)in_sizes; (void)n_in; (void)out_size; (void)ws_size;
  const float* x   = (const float*)d_in[0];
  const float* pos = (const float*)d_in[1];
  const int*   ei  = (const int*)d_in[2];
  const float* ep  = (const float*)d_in[3];
  float* out = (float*)d_out;
  int*   ws  = (int*)d_ws;

  int*   rowcnt   = ws + WS_ROWCNT;
  int*   cursor   = ws + WS_CURSOR;
  int*   hist1    = ws + WS_HIST1;
  int*   h2a      = ws + WS_HIST2A;
  int*   h2b      = ws + WS_HIST2B;
  int*   scal     = ws + WS_SCAL;
  int*   rowstart = ws + WS_ROWSTART;
  int*   csrcol   = ws + WS_CSRCOL;
  float* csrval   = (float*)(ws + WS_CSRVAL);
  int*   topi     = ws + WS_TOPI;
  float* topv     = (float*)(ws + WS_TOPV);

  hipMemsetAsync(ws, 0, (size_t)WS_ZERO_N * sizeof(int), stream);  // counters/histograms
  count_kernel  <<<(EE + 255) / 256, 256, 0, stream>>>(ei, rowcnt);
  scan_kernel   <<<1, 256, 0, stream>>>(rowcnt, rowstart);
  scatter_kernel<<<(EE + 255) / 256, 256, 0, stream>>>(ei, ep, rowstart, cursor, csrcol, csrval);
  topk_kernel   <<<(NN + 255) / 256, 256, 0, stream>>>(rowstart, csrcol, csrval, topi, topv);
  tri_kernel    <<<NN / WAVES, 32 * WAVES, 0, stream>>>(x, pos, rowstart, csrcol, csrval,
                                                        topi, topv, out);
  hist1_kernel  <<<(NTRI + 255) / 256, 256, 0, stream>>>(out, hist1);
  qscan1_kernel <<<1, 256, 0, stream>>>(hist1, scal);
  hist2_kernel  <<<(NTRI + 255) / 256, 256, 0, stream>>>(out, scal, h2a, h2b);
  qscan2_kernel <<<1, 256, 0, stream>>>(h2a, h2b, scal);
  keep_kernel   <<<(NTRI + 255) / 256, 256, 0, stream>>>(scal, out);
}
